// FullAttention_19791209300794
// MI455X (gfx1250) — compile-verified, matrix operand order checked by reading, not run on hardware
//
#include <hip/hip_runtime.h>

typedef __attribute__((ext_vector_type(16))) __bf16 v16bf;
typedef __attribute__((ext_vector_type(8)))  __bf16 v8bf;
typedef __attribute__((ext_vector_type(8)))  float  v8f;
typedef __attribute__((ext_vector_type(4)))  unsigned v4u;
typedef __attribute__((ext_vector_type(8)))  int      v8i;
typedef __attribute__((ext_vector_type(4)))  int      v4i;

// Problem sizes (fixed by the reference)
constexpr int Bsz = 4, Lseq = 2048, Cdim = 1024, Hn = 16, Dh = 64;
constexpr int Mrows = Bsz * Lseq;          // 8192
constexpr int Nqkv  = 3 * Cdim;            // 3072

// ---------------------------------------------------------------------------
// CDNA5 feature detection (device pass only; fallbacks keep compile safe)
// ---------------------------------------------------------------------------
#if defined(__AMDGCN__) && __has_builtin(__builtin_amdgcn_global_load_async_to_lds_b128) && __has_builtin(__builtin_amdgcn_s_wait_asynccnt)
#define HAS_ASYNC_LDS 1
#else
#define HAS_ASYNC_LDS 0
#endif

#if defined(__AMDGCN__) && __has_builtin(__builtin_amdgcn_tensor_load_to_lds) && __has_builtin(__builtin_amdgcn_s_wait_tensorcnt)
#define HAS_TDM 1
#else
#define HAS_TDM 0
#endif

#if defined(__AMDGCN__)
#define LDS_OFFSET(p) ((unsigned)(unsigned long long)(__attribute__((address_space(3))) void*)(p))
#endif

#if HAS_ASYNC_LDS
// async 16B global->LDS copy (b128 expressed as 4 x i32 per builtin prototype)
__device__ __forceinline__ void async_copy_b128(const void* g, void* l) {
  __builtin_amdgcn_global_load_async_to_lds_b128(
      (__attribute__((address_space(1))) v4i*)g,
      (__attribute__((address_space(3))) v4i*)l, 0, 0);
}
#endif

#if HAS_TDM
// 2D tile DMA global->LDS via Tensor Data Mover.
// D# group0/group1 packed per CDNA5 ISA 8.3/8.4.  padcfg carries
// data_size / pad_enable / pad_interval / pad_amount bits of group1 word0.
__device__ __forceinline__ void tdm_load_2d(const void* gaddr, unsigned lds_off,
                                            unsigned dim0, unsigned dim1,
                                            unsigned stride0,
                                            unsigned tile0, unsigned tile1,
                                            unsigned padcfg) {
  unsigned long long ga = (unsigned long long)gaddr;
  v4u g0;
  g0[0] = 1u;                                            // count=1, no gather
  g0[1] = lds_off;                                       // lds_addr
  g0[2] = (unsigned)ga;                                  // global_addr[31:0]
  g0[3] = (unsigned)((ga >> 32) & 0x1FFFFFFu) | 0x80000000u; // addr[56:32]|type=2
  v8i g1;
  g1[0] = (int)padcfg;                                   // mask=0, data_size, pad
  g1[1] = (int)((dim0 & 0xffffu) << 16);                 // [47:32]=0, dim0.lo
  g1[2] = (int)(((dim0 >> 16) & 0xffffu) | ((dim1 & 0xffffu) << 16));
  g1[3] = (int)(((dim1 >> 16) & 0xffffu) | ((tile0 & 0xffffu) << 16));
  g1[4] = (int)(tile1 & 0xffffu);                        // tile_dim1, tile_dim2=0
  g1[5] = (int)stride0;                                  // dim0_stride[31:0]
  g1[6] = 0;                                             // stride hi, dim1_stride
  g1[7] = 0;
  v4i z4 = {0, 0, 0, 0};
#if __clang_major__ >= 23
  v8i z8 = {0, 0, 0, 0, 0, 0, 0, 0};
  __builtin_amdgcn_tensor_load_to_lds(g0, g1, z4, z4, z8, 0);
#else
  __builtin_amdgcn_tensor_load_to_lds(g0, g1, z4, z4, 0);
#endif
}
// data_size=2B(code1), pad_enable, pad_interval: 32 DWORDs(code4),
// pad_amount: 4 DWORDs(code3)  -> LDS row stride 64+8 elements = 72
#define TDM_PAD_2B_32DW_4DW ((1u << 16) | (1u << 20) | (4u << 22) | (3u << 25))
#endif

// ---------------------------------------------------------------------------
// Fragment loader: CDNA5 16-bit A/B WMMA layout (wave32).
// lane: row/col = lane&15 ; kb=(lane>>4)*8 ; elems 0-7 -> K=kb..kb+7,
// elems 8-15 -> K=16+kb..16+kb+7.  p points at the lane's row start.
// ---------------------------------------------------------------------------
__device__ __forceinline__ v16bf load_frag(const __bf16* p, int kb) {
  union { v16bf v; v8bf h[2]; } u;
  u.h[0] = *(const v8bf*)(p + kb);
  u.h[1] = *(const v8bf*)(p + kb + 16);
  return u.v;
}

// ---------------------------------------------------------------------------
// Elementwise fp32 -> bf16 (8 elems / thread)
// ---------------------------------------------------------------------------
__global__ void f32_to_bf16_kernel(const float* __restrict__ in,
                                   __bf16* __restrict__ out, int n8) {
  int i = blockIdx.x * blockDim.x + threadIdx.x;
  if (i >= n8) return;
  const float4 a = ((const float4*)in)[2 * i];
  const float4 b = ((const float4*)in)[2 * i + 1];
  v8bf o;
  o[0] = (__bf16)a.x; o[1] = (__bf16)a.y; o[2] = (__bf16)a.z; o[3] = (__bf16)a.w;
  o[4] = (__bf16)b.x; o[5] = (__bf16)b.y; o[6] = (__bf16)b.z; o[7] = (__bf16)b.w;
  ((v8bf*)out)[i] = o;
}

// ---------------------------------------------------------------------------
// bf16 GEMM, f32 accumulate.  C[M,N] = A[M,K] * B[K,N]  (all row-major).
// Block tile 128x128x32, 8 waves, each wave 32(M) x 64(N) = 2x4 WMMA tiles.
// A tile staged via async global->LDS (no VGPR round trip); B staged with an
// in-LDS transpose to N-major so both fragment loads are two ds_load_b128.
// ---------------------------------------------------------------------------
constexpr int ASTR = 40;   // padded LDS stride (elements), multiple of 8
constexpr int BSTR = 40;

__global__ __launch_bounds__(256)
void gemm_bf16_wmma(const __bf16* __restrict__ A, const __bf16* __restrict__ Bm,
                    float* __restrict__ Cm, int M, int N, int K) {
  __shared__ __bf16 As[128 * ASTR];
  __shared__ __bf16 Bs[128 * BSTR];

  const int t    = threadIdx.x;
  const int lane = t & 31;
  const int wv   = t >> 5;        // 0..7
  const int wm   = wv & 3;        // 4 waves along M
  const int wn   = wv >> 2;       // 2 waves along N
  const int m0   = blockIdx.y * 128;
  const int n0   = blockIdx.x * 128;
  const int waveM = wm * 32;
  const int waveN = wn * 64;
  const int hrow = lane & 15;
  const int kb   = (lane >> 4) * 8;

  v8f acc[2][4];
#pragma unroll
  for (int i = 0; i < 2; i++)
#pragma unroll
    for (int j = 0; j < 4; j++) acc[i][j] = {};

  for (int k0 = 0; k0 < K; k0 += 32) {
    // ---- stage A tile 128x32 (straight copy, row-major) ----
#if HAS_ASYNC_LDS
#pragma unroll
    for (int i = 0; i < 2; i++) {
      int lin = t + i * 256;
      int r   = lin >> 2;
      int c8  = (lin & 3) * 8;
      async_copy_b128(&A[(size_t)(m0 + r) * K + k0 + c8], &As[r * ASTR + c8]);
    }
#else
    v8bf areg[2];
    int  adst[2];
#pragma unroll
    for (int i = 0; i < 2; i++) {
      int lin = t + i * 256;
      int r   = lin >> 2;
      int c8  = (lin & 3) * 8;
      areg[i] = *(const v8bf*)(&A[(size_t)(m0 + r) * K + k0 + c8]);
      adst[i] = r * ASTR + c8;
    }
#pragma unroll
    for (int i = 0; i < 2; i++) *(v8bf*)(&As[adst[i]]) = areg[i];
#endif
    // ---- stage B tile 32x128 transposed -> Bs[n][k] (loads first, then scatter) ----
    {
      int nc = t & 7;           // n chunk of 16
      int kk = t >> 3;          // 0..31
      const __bf16* src = &Bm[(size_t)(k0 + kk) * N + n0 + nc * 16];
      v8bf b0 = *(const v8bf*)(src);
      v8bf b1 = *(const v8bf*)(src + 8);
#pragma unroll
      for (int j = 0; j < 8; j++) Bs[(nc * 16 + j) * BSTR + kk] = b0[j];
#pragma unroll
      for (int j = 0; j < 8; j++) Bs[(nc * 16 + 8 + j) * BSTR + kk] = b1[j];
    }
    if (k0 + 32 < K) {   // hint next tiles into cache
      __builtin_prefetch(&A[(size_t)(m0 + (t >> 1)) * K + k0 + 32], 0, 1);
      __builtin_prefetch(&Bm[(size_t)(k0 + 32 + (t >> 3)) * N + n0 + (t & 7) * 16], 0, 1);
    }
#if HAS_ASYNC_LDS
    __builtin_amdgcn_s_wait_asynccnt(0);
#endif
    __syncthreads();

    v16bf afr[2], bfr[4];
#pragma unroll
    for (int i = 0; i < 2; i++)
      afr[i] = load_frag(&As[(waveM + i * 16 + hrow) * ASTR], kb);
#pragma unroll
    for (int j = 0; j < 4; j++)
      bfr[j] = load_frag(&Bs[(waveN + j * 16 + hrow) * BSTR], kb);

#pragma unroll
    for (int i = 0; i < 2; i++)
#pragma unroll
      for (int j = 0; j < 4; j++)
        acc[i][j] = __builtin_amdgcn_wmma_f32_16x16x32_bf16(
            false, afr[i], false, bfr[j], (short)0, acc[i][j], false, false);
    __syncthreads();
  }

  // ---- store: C layout lane = N&15, vgpr v -> row v + 8*(lane>>4) ----
  const int hl = lane >> 4;
#pragma unroll
  for (int i = 0; i < 2; i++)
#pragma unroll
    for (int j = 0; j < 4; j++)
#pragma unroll
      for (int v = 0; v < 8; v++) {
        int row = m0 + waveM + i * 16 + v + 8 * hl;
        int col = n0 + waveN + j * 16 + hrow;
        Cm[(size_t)row * N + col] = acc[i][j][v];
      }
}

// ---------------------------------------------------------------------------
// RMSNorm + RoPE epilogue. One wave per (b,l,h); lane owns d and d+32.
// qkv: [B*L][3*C] f32 ; outputs Q,K,V: [B*H][L][D] bf16.
// ---------------------------------------------------------------------------
__global__ __launch_bounds__(256)
void qkv_post_kernel(const float* __restrict__ qkv, const float* __restrict__ qg,
                     const float* __restrict__ kg, __bf16* __restrict__ Q,
                     __bf16* __restrict__ Kd, __bf16* __restrict__ V) {
  int w    = (blockIdx.x * blockDim.x + threadIdx.x) >> 5;
  int lane = threadIdx.x & 31;
  int h = w % Hn;
  int l = (w / Hn) % Lseq;
  int b = w / (Hn * Lseq);

  const float* rowp = qkv + ((size_t)(b * Lseq + l)) * Nqkv + h * Dh;
  float q1 = rowp[lane],        q2 = rowp[lane + 32];
  float k1 = rowp[Cdim + lane], k2 = rowp[Cdim + lane + 32];
  float v1 = rowp[2 * Cdim + lane], v2 = rowp[2 * Cdim + lane + 32];

  float sq = q1 * q1 + q2 * q2;
  float sk = k1 * k1 + k2 * k2;
#pragma unroll
  for (int m = 1; m < 32; m <<= 1) {
    sq += __shfl_xor(sq, m, 32);
    sk += __shfl_xor(sk, m, 32);
  }
  float rq = rsqrtf(sq * (1.0f / 64.0f) + 1e-6f);
  float rk = rsqrtf(sk * (1.0f / 64.0f) + 1e-6f);
  float g1 = qg[lane], g2 = qg[lane + 32];
  float e1 = kg[lane], e2 = kg[lane + 32];
  q1 *= rq * g1; q2 *= rq * g2;
  k1 *= rk * e1; k2 *= rk * e2;

  float invf = __powf(10000.0f, -(float)lane * (1.0f / 32.0f));
  float ang  = (float)l * invf;
  float c = cosf(ang), s = sinf(ang);
  float qo1 = q1 * c - q2 * s, qo2 = q2 * c + q1 * s;
  float ko1 = k1 * c - k2 * s, ko2 = k2 * c + k1 * s;

  size_t ob = ((size_t)(b * Hn + h) * Lseq + l) * Dh;
  Q[ob + lane]       = (__bf16)qo1;  Q[ob + lane + 32] = (__bf16)qo2;
  Kd[ob + lane]      = (__bf16)ko1;  Kd[ob + lane + 32] = (__bf16)ko2;
  V[ob + lane]       = (__bf16)v1;   V[ob + lane + 32]  = (__bf16)v2;
}

// ---------------------------------------------------------------------------
// Flash attention. Grid: (L/128, B*H). 8 waves x 16 query rows.
// K chunk staged once per block (TDM DMA when available); V transposed in LDS.
// Q,K,V: [B*H][L][D] bf16 -> O: [B*L][C] bf16.
// ---------------------------------------------------------------------------
constexpr int KSTR = 72;   // 64 + 8 pad, matches TDM pad config
constexpr int VSTR = 72;
constexpr int PSTR = 72;

__global__ __launch_bounds__(256)
void attn_kernel(const __bf16* __restrict__ Q, const __bf16* __restrict__ Kd,
                 const __bf16* __restrict__ V, __bf16* __restrict__ O) {
  __shared__ __bf16 Ks[64 * KSTR];          // K chunk [key][d]
  __shared__ __bf16 Vt[Dh * VSTR];          // V chunk transposed [d][key]
  __shared__ __bf16 Ps[8 * 16 * PSTR];      // per-wave P tile [m][key]

  const int t    = threadIdx.x;
  const int lane = t & 31;
  const int wv   = t >> 5;
  const int hrow = lane & 15;
  const int hl   = lane >> 4;
  const int kb   = hl * 8;
  const int bh   = blockIdx.y;
  const int q0   = blockIdx.x * 128 + wv * 16;
  const size_t base = (size_t)bh * Lseq * Dh;
  const float scale = 0.125f;               // D^-0.5

  // Q fragments (persistent): A-layout, two K-halves of D=64
  v16bf qf[2];
  {
    const __bf16* qp = Q + base + (size_t)(q0 + hrow) * Dh;
#pragma unroll
    for (int tt = 0; tt < 2; tt++) qf[tt] = load_frag(qp + tt * 32, kb);
  }

  float mstate[8], lstate[8];
  v8f oacc[4];
#pragma unroll
  for (int v = 0; v < 8; v++) { mstate[v] = -1e30f; lstate[v] = 0.0f; }
#pragma unroll
  for (int dt = 0; dt < 4; dt++) oacc[dt] = {};

  __bf16* pw = &Ps[wv * 16 * PSTR];

  for (int key0 = 0; key0 < Lseq; key0 += 64) {
    // ---- stage K chunk [64 keys][64 d] ----
#if HAS_TDM
    if (wv == 0) {
      tdm_load_2d(Kd + base + (size_t)key0 * Dh, LDS_OFFSET(Ks),
                  /*dim0*/Dh, /*dim1*/Lseq, /*stride0*/Dh,
                  /*tile0*/Dh, /*tile1*/64, TDM_PAD_2B_32DW_4DW);
      __builtin_amdgcn_s_wait_tensorcnt(0);
    }
#else
    {
      int key = t >> 2, dc = (t & 3) * 16;
      const __bf16* kp = Kd + base + (size_t)(key0 + key) * Dh + dc;
      v8bf c0 = *(const v8bf*)kp;
      v8bf c1 = *(const v8bf*)(kp + 8);
      *(v8bf*)&Ks[key * KSTR + dc]     = c0;
      *(v8bf*)&Ks[key * KSTR + dc + 8] = c1;
    }
#endif
    // ---- stage V chunk transposed: Vt[d][key] ----
    {
      int key = t >> 2;
      int dc  = (t & 3) * 16;
      const __bf16* vp = V + base + (size_t)(key0 + key) * Dh + dc;
      v8bf a0 = *(const v8bf*)vp;
      v8bf a1 = *(const v8bf*)(vp + 8);
#pragma unroll
      for (int j = 0; j < 8; j++) Vt[(dc + j) * VSTR + key]     = a0[j];
#pragma unroll
      for (int j = 0; j < 8; j++) Vt[(dc + 8 + j) * VSTR + key] = a1[j];
    }
    __syncthreads();

    // ---- scores S = Q K^T for 4 key tiles of 16 (K frags from LDS) ----
    v8f sa[4];
#pragma unroll
    for (int kt = 0; kt < 4; kt++) {
      v8f sacc = {};
      const __bf16* kp = &Ks[(kt * 16 + hrow) * KSTR];
#pragma unroll
      for (int tt = 0; tt < 2; tt++) {
        v16bf kf = load_frag(kp + tt * 32, kb);
        sacc = __builtin_amdgcn_wmma_f32_16x16x32_bf16(
            false, qf[tt], false, kf, (short)0, sacc, false, false);
      }
      sa[kt] = sacc * scale;
    }

    // ---- online softmax per row (row v+8*hl lives in the lane's 16-lane half) ----
#pragma unroll
    for (int v = 0; v < 8; v++) {
      float mx = -1e30f;
#pragma unroll
      for (int kt = 0; kt < 4; kt++) mx = fmaxf(mx, sa[kt][v]);
#pragma unroll
      for (int m = 1; m < 16; m <<= 1) mx = fmaxf(mx, __shfl_xor(mx, m, 32));
      float mn = fmaxf(mstate[v], mx);
      float al = __expf(mstate[v] - mn);
      float rs = 0.0f;
#pragma unroll
      for (int kt = 0; kt < 4; kt++) {
        float p = __expf(sa[kt][v] - mn);
        sa[kt][v] = p;
        rs += p;
      }
#pragma unroll
      for (int m = 1; m < 16; m <<= 1) rs += __shfl_xor(rs, m, 32);
      lstate[v] = lstate[v] * al + rs;
      mstate[v] = mn;
#pragma unroll
      for (int dt = 0; dt < 4; dt++) oacc[dt][v] *= al;
    }

    // ---- P (C-layout) -> LDS, re-read as A-layout fragments ----
#pragma unroll
    for (int kt = 0; kt < 4; kt++)
#pragma unroll
      for (int v = 0; v < 8; v++)
        pw[(v + 8 * hl) * PSTR + kt * 16 + hrow] = (__bf16)sa[kt][v];

    v16bf pf[2];
#pragma unroll
    for (int tt = 0; tt < 2; tt++)
      pf[tt] = load_frag(&pw[hrow * PSTR] + tt * 32, kb);

    // ---- O += P @ V  (V frags from transposed LDS) ----
#pragma unroll
    for (int dt = 0; dt < 4; dt++) {
      const __bf16* vp2 = &Vt[(dt * 16 + hrow) * VSTR];
#pragma unroll
      for (int tt = 0; tt < 2; tt++) {
        v16bf vf = load_frag(vp2 + tt * 32, kb);
        oacc[dt] = __builtin_amdgcn_wmma_f32_16x16x32_bf16(
            false, pf[tt], false, vf, (short)0, oacc[dt], false, false);
      }
    }
    __syncthreads();   // before next chunk overwrites Ks/Vt
  }

  // ---- normalize + store O as [B*L][C] bf16 ----
  const int b = bh >> 4;
  const int h = bh & 15;
#pragma unroll
  for (int v = 0; v < 8; v++) {
    float inv = 1.0f / lstate[v];
    int l = q0 + v + 8 * hl;
    size_t rowb = ((size_t)(b * Lseq + l)) * Cdim + h * Dh;
#pragma unroll
    for (int dt = 0; dt < 4; dt++)
      O[rowb + dt * 16 + hrow] = (__bf16)(oacc[dt][v] * inv);
  }
}

// ---------------------------------------------------------------------------
// Host launcher
// ---------------------------------------------------------------------------
extern "C" void kernel_launch(void* const* d_in, const int* in_sizes, int n_in,
                              void* d_out, int out_size, void* d_ws, size_t ws_size,
                              hipStream_t stream) {
  const float* x       = (const float*)d_in[0];
  const float* w_qkv   = (const float*)d_in[1];
  const float* w_out   = (const float*)d_in[2];
  const float* q_gamma = (const float*)d_in[3];
  const float* k_gamma = (const float*)d_in[4];
  float* out = (float*)d_out;

  // workspace layout
  char* ws = (char*)d_ws;
  __bf16* Xb   = (__bf16*)ws; ws += (size_t)Mrows * Cdim * 2;
  __bf16* Wqb  = (__bf16*)ws; ws += (size_t)Cdim * Nqkv * 2;
  __bf16* Wob  = (__bf16*)ws; ws += (size_t)Cdim * Cdim * 2;
  float*  qkv  = (float*) ws; ws += (size_t)Mrows * Nqkv * 4;
  __bf16* Qb   = (__bf16*)ws; ws += (size_t)Mrows * Cdim * 2;
  __bf16* Kb   = (__bf16*)ws; ws += (size_t)Mrows * Cdim * 2;
  __bf16* Vb   = (__bf16*)ws; ws += (size_t)Mrows * Cdim * 2;
  __bf16* Ob   = (__bf16*)ws; ws += (size_t)Mrows * Cdim * 2;

  // 1) convert to bf16
  {
    int n8 = Mrows * Cdim / 8;
    f32_to_bf16_kernel<<<(n8 + 255) / 256, 256, 0, stream>>>(x, Xb, n8);
    n8 = Cdim * Nqkv / 8;
    f32_to_bf16_kernel<<<(n8 + 255) / 256, 256, 0, stream>>>(w_qkv, Wqb, n8);
    n8 = Cdim * Cdim / 8;
    f32_to_bf16_kernel<<<(n8 + 255) / 256, 256, 0, stream>>>(w_out, Wob, n8);
  }
  // 2) QKV projection GEMM: [8192,1024] x [1024,3072] -> f32
  {
    dim3 grid(Nqkv / 128, Mrows / 128);
    gemm_bf16_wmma<<<grid, 256, 0, stream>>>(Xb, Wqb, qkv, Mrows, Nqkv, Cdim);
  }
  // 3) RMSNorm + RoPE -> bf16 Q/K/V in [B*H][L][D]
  {
    int waves = Bsz * Lseq * Hn;               // 131072
    qkv_post_kernel<<<waves / 8, 256, 0, stream>>>(qkv, q_gamma, k_gamma, Qb, Kb, Vb);
  }
  // 4) flash attention -> Ob [B*L][C] bf16
  {
    dim3 grid(Lseq / 128, Bsz * Hn);
    attn_kernel<<<grid, 256, 0, stream>>>(Qb, Kb, Vb, Ob);
  }
  // 5) output projection GEMM: [8192,1024] x [1024,1024] -> f32 out
  {
    dim3 grid(Cdim / 128, Mrows / 128);
    gemm_bf16_wmma<<<grid, 256, 0, stream>>>(Ob, Wob, out, Mrows, Cdim, Cdim);
  }
}
